// DynamicGraphIpaFrameDenoiser_56856777064493
// MI455X (gfx1250) — matrix operand order repaired and checked by reading, not compiled, not run on hardware
//
#include <hip/hip_runtime.h>
#include <cstdint>
#include <cstddef>

typedef unsigned short u16;
typedef __attribute__((ext_vector_type(16))) __bf16 v16bf;
typedef __attribute__((ext_vector_type(8)))  __bf16 v8bf;
typedef __attribute__((ext_vector_type(8)))  float  v8f;

// ---------------- model dims ----------------
#define NN    1024
#define EE    32768
#define ESS   8192
#define CS    256
#define CL    128
#define CZ    128
#define HH    8
#define PQ    8
#define PV    12
#define ERAW  203
#define EIN   971
#define EINP  992        // EIN padded to multiple of 32
#define ETH   384
#define FEATS 1920       // H*(CH + PV*3 + PV + CZ)

// ---------------- device helpers ----------------
__device__ __forceinline__ u16 f2bf(float f) {
  unsigned u = __float_as_uint(f);
  u += 0x7fffu + ((u >> 16) & 1u);
  return (u16)(u >> 16);
}

__device__ __forceinline__ void atomicMaxF(float* addr, float val) {
  int* ia = (int*)addr;
  int old = __float_as_int(*addr);
  while (__int_as_float(old) < val) {
    int assumed = old;
    old = atomicCAS(ia, assumed, __float_as_int(val));
    if (old == assumed) break;
  }
}

// LDS byte offset of a pointer into a __shared__ array (ptrtoint of AS(3) ptr)
__device__ __forceinline__ unsigned lds_off(const u16* p) {
  return (unsigned)(unsigned long long)(__attribute__((address_space(3))) const u16*)p;
}

// CDNA5 async global -> LDS copy, 16 bytes per lane, tracked by ASYNCcnt.
__device__ __forceinline__ void async_copy_b128(unsigned lds_addr, const u16* g) {
  asm volatile("global_load_async_to_lds_b128 %0, %1, off"
               :: "v"(lds_addr), "v"((unsigned long long)(size_t)g)
               : "memory");
}
__device__ __forceinline__ void wait_async0() {
  asm volatile("s_wait_asynccnt 0x0" ::: "memory");
}

// ---------------- WMMA GEMM ----------------
// C[M x Ncp] = A[M x K] (bf16, row major, stride K) @ W[K x Ncp] (bf16, row major)
// + bias (first nbias cols). act: 0 = none, 1 = relu.
// Block = 128 threads (4 waves). Block tile: 32 (M) x 128 (N); each wave owns a
// 32-wide N strip (two 16-wide WMMA strips) over two 16-row M sub-tiles:
// 4 x v_wmma_f32_16x16x32_bf16 per wave per K-step.
// A tile (32x32 bf16) is double-buffered in LDS via async global->LDS copies.
// grid = (M/32, ceil(Ncp/128)).
__global__ __launch_bounds__(128)
void k_gemm(const u16* __restrict__ A, const u16* __restrict__ W,
            const float* __restrict__ bias, int nbias,
            float* __restrict__ outf, u16* __restrict__ outb,
            int K, int Ncp, int act) {
  __shared__ alignas(128) u16 lds_a[2][32 * 32];
  const int tid = threadIdx.x;
  const int lane = tid & 31;
  const int wave = tid >> 5;
  const int m0 = blockIdx.x << 5;                      // 32 rows per block
  const int n0 = (blockIdx.y << 7) + (wave << 5);      // 32 cols per wave
  const bool valid0 = (n0 < Ncp);
  const bool valid1 = (n0 + 16 < Ncp);
  const int g = lane >> 4;
  const int mr = lane & 15;

  // staging: thread t copies 16B: row = t>>2 (0..31), col = (t&3)*8
  const int sr = tid >> 2;
  const int sc = (tid & 3) << 3;
  const u16* aptr = A + (size_t)(m0 + sr) * K + sc;
  const unsigned dst0 = lds_off(&lds_a[0][sr * 32 + sc]);
  const unsigned dst1 = lds_off(&lds_a[1][sr * 32 + sc]);

  const u16* wptr = W + n0 + (lane & 15);              // col = n0 + lane%16
  const size_t wstep = (size_t)32 * Ncp;

  v8f acc00 = {0.f,0.f,0.f,0.f,0.f,0.f,0.f,0.f};
  v8f acc01 = acc00, acc10 = acc00, acc11 = acc00;

  async_copy_b128(dst0, aptr);                          // prologue: tile 0
  int cur = 0;
  for (int kk = 0; kk < K; kk += 32) {
    wait_async0();                                      // my writes to buf[cur] landed
    __syncthreads();                                    // everyone's landed; prev reads done
    if (kk + 32 < K)
      async_copy_b128(cur ? dst0 : dst1, aptr + 32);    // stream next tile into other buf
    if (valid0) {
      const u16* base = cur ? &lds_a[1][0] : &lds_a[0][0];
      // A fragments per CDNA5 16-bit 16x32 layout:
      // elems 0..7 -> K = g*8 + j ; elems 8..15 -> K = 16 + g*8 + j
      const u16* r0 = base + (mr << 5);
      v8bf lo0 = *(const v8bf*)&r0[g << 3];
      v8bf hi0 = *(const v8bf*)&r0[16 + (g << 3)];
      v16bf a0 = __builtin_shufflevector(lo0, hi0, 0,1,2,3,4,5,6,7,8,9,10,11,12,13,14,15);
      const u16* r1 = base + ((16 + mr) << 5);
      v8bf lo1 = *(const v8bf*)&r1[g << 3];
      v8bf hi1 = *(const v8bf*)&r1[16 + (g << 3)];
      v16bf a1 = __builtin_shufflevector(lo1, hi1, 0,1,2,3,4,5,6,7,8,9,10,11,12,13,14,15);
      // B fragments: lane%16 = N col, elems in same K order as A
      v16bf b0;
      for (int s = 0; s < 16; ++s)
        b0[s] = *(const __bf16*)&wptr[(size_t)(((s >> 3) << 4) + (g << 3) + (s & 7)) * Ncp];
      if (kk + 32 < K)
        __builtin_prefetch(wptr + wstep, 0, 1);
      acc00 = __builtin_amdgcn_wmma_f32_16x16x32_bf16(false, a0, false, b0, (short)0, acc00, false, false);
      acc10 = __builtin_amdgcn_wmma_f32_16x16x32_bf16(false, a1, false, b0, (short)0, acc10, false, false);
      if (valid1) {
        v16bf b1;
        for (int s = 0; s < 16; ++s)
          b1[s] = *(const __bf16*)&wptr[16 + (size_t)(((s >> 3) << 4) + (g << 3) + (s & 7)) * Ncp];
        acc01 = __builtin_amdgcn_wmma_f32_16x16x32_bf16(false, a0, false, b1, (short)0, acc01, false, false);
        acc11 = __builtin_amdgcn_wmma_f32_16x16x32_bf16(false, a1, false, b1, (short)0, acc11, false, false);
      }
    }
    aptr += 32;
    wptr += wstep;
    cur ^= 1;
  }

  const int nc = lane & 15;
  const int mb = m0 + ((lane >> 4) << 3);
  for (int strip = 0; strip < 2; ++strip) {
    if (strip == 0 ? !valid0 : !valid1) continue;
    const int n = n0 + strip * 16 + nc;
    const float bv = (bias && n < nbias) ? bias[n] : 0.f;
    v8f aco0 = strip ? acc01 : acc00;
    v8f aco1 = strip ? acc11 : acc10;
    for (int v = 0; v < 8; ++v) {
      float x0 = aco0[v] + bv;
      float x1 = aco1[v] + bv;
      if (act == 1) { x0 = fmaxf(x0, 0.f); x1 = fmaxf(x1, 0.f); }
      size_t off0 = (size_t)(mb + v) * Ncp + n;
      size_t off1 = (size_t)(mb + 16 + v) * Ncp + n;
      if (outf) { outf[off0] = x0; outf[off1] = x1; }
      if (outb) { outb[off0] = f2bf(x0); outb[off1] = f2bf(x1); }
    }
  }
}

// weight f32[K x Nc] -> bf16[Kp x Ncp], zero padded
__global__ void k_w2bf(const float* __restrict__ w, u16* __restrict__ dst,
                       int K, int Kp, int Nc, int Ncp) {
  int i = blockIdx.x * blockDim.x + threadIdx.x;
  if (i >= Kp * Ncp) return;
  int r = i / Ncp, c = i % Ncp;
  float v = (r < K && c < Nc) ? w[(size_t)r * Nc + c] : 0.f;
  dst[i] = f2bf(v);
}

// ---------------- elementwise ----------------
__global__ void k_copy_dual(float* __restrict__ f, u16* __restrict__ b,
                            const float* __restrict__ src, int n) {
  int i = blockIdx.x * blockDim.x + threadIdx.x;
  if (i >= n) return;
  float v = src[i];
  f[i] = v;
  if (b) b[i] = f2bf(v);
}
__global__ void k_fill(float* __restrict__ p, float v, int n) {
  int i = blockIdx.x * blockDim.x + threadIdx.x;
  if (i < n) p[i] = v;
}
__global__ void k_sigmul(float* __restrict__ out, const float* __restrict__ g,
                         const float* __restrict__ l, int n) {
  int i = blockIdx.x * blockDim.x + threadIdx.x;
  if (i >= n) return;
  out[i] = l[i] / (1.f + expf(-g[i]));
}
__global__ void k_addcvt(u16* __restrict__ dst, const float* __restrict__ a,
                         const float* __restrict__ b, int n) {
  int i = blockIdx.x * blockDim.x + threadIdx.x;
  if (i < n) dst[i] = f2bf(a[i] + b[i]);
}
__global__ void k_add_rowscale(float* __restrict__ out, const float* __restrict__ add,
                               const float* __restrict__ m, int rows, int D) {
  int i = blockIdx.x * blockDim.x + threadIdx.x;
  if (i >= rows * D) return;
  out[i] += add[i] * m[i / D];
}
__global__ void k_rowscale_dual(float* __restrict__ x, u16* __restrict__ xb,
                                const float* __restrict__ m, int rows, int D) {
  int i = blockIdx.x * blockDim.x + threadIdx.x;
  if (i >= rows * D) return;
  float v = x[i] * m[i / D];
  x[i] = v;
  if (xb) xb[i] = f2bf(v);
}

// ---------------- layernorm (one row per block) ----------------
__global__ __launch_bounds__(128)
void k_layernorm(const float* __restrict__ x, const float* __restrict__ res,
                 const float* __restrict__ rs, const float* __restrict__ g,
                 const float* __restrict__ b, float* __restrict__ outf,
                 u16* __restrict__ outb, int D) {
  const int row = blockIdx.x;
  const float sc = rs ? rs[row] : 1.f;
  const float* xr = x + (size_t)row * D;
  const float* rr = res ? res + (size_t)row * D : nullptr;
  __shared__ float sh1[128], sh2[128];
  float s1 = 0.f, s2 = 0.f;
  for (int i = threadIdx.x; i < D; i += 128) {
    float v = xr[i] + (rr ? rr[i] * sc : 0.f);
    s1 += v; s2 += v * v;
  }
  sh1[threadIdx.x] = s1; sh2[threadIdx.x] = s2;
  __syncthreads();
  for (int s = 64; s > 0; s >>= 1) {
    if (threadIdx.x < s) { sh1[threadIdx.x] += sh1[threadIdx.x + s];
                           sh2[threadIdx.x] += sh2[threadIdx.x + s]; }
    __syncthreads();
  }
  const float mu = sh1[0] / D;
  const float inv = rsqrtf(sh2[0] / D - mu * mu + 1e-5f);
  float* of = outf + (size_t)row * D;
  u16* ob = outb ? outb + (size_t)row * D : nullptr;
  for (int i = threadIdx.x; i < D; i += 128) {
    float v = xr[i] + (rr ? rr[i] * sc : 0.f);
    float o = (v - mu) * inv * g[i] + b[i];
    of[i] = o;
    if (ob) ob[i] = f2bf(o);
  }
}

// ---------------- gather / concat ----------------
__global__ __launch_bounds__(128)
void k_gather_edge(u16* __restrict__ out, const float* __restrict__ raw,
                   const float* __restrict__ s, const float* __restrict__ lat,
                   const int* __restrict__ ei, int Ecnt) {
  const int e = blockIdx.x;
  const int src = ei[e], dst = ei[Ecnt + e];
  u16* o = out + (size_t)e * EINP;
  const float* r0 = raw + (size_t)e * ERAW;
  const float* ss = s + (size_t)src * CS;
  const float* sd = s + (size_t)dst * CS;
  const float* ls = lat + (size_t)src * CL;
  const float* ld = lat + (size_t)dst * CL;
  for (int i = threadIdx.x; i < EINP; i += 128) {
    float v;
    if (i < ERAW) v = r0[i];
    else if (i < ERAW + CS) v = ss[i - ERAW];
    else if (i < ERAW + 2 * CS) v = sd[i - ERAW - CS];
    else if (i < ERAW + 2 * CS + CL) v = ls[i - ERAW - 2 * CS];
    else if (i < EIN) v = ld[i - ERAW - 2 * CS - CL];
    else v = 0.f;
    o[i] = f2bf(v);
  }
}

__global__ void k_concat_ls(u16* __restrict__ out, const float* __restrict__ lat,
                            const float* __restrict__ s, const float* __restrict__ mask,
                            int rows) {
  int i = blockIdx.x * blockDim.x + threadIdx.x;
  if (i >= rows * 384) return;
  int r = i / 384, c = i % 384;
  float m = mask ? mask[r] : 1.f;
  float v = (c < CL) ? lat[(size_t)r * CL + c] : s[(size_t)r * CS + (c - CL)];
  out[i] = f2bf(v * m);
}

__global__ void k_concat_et(float* __restrict__ outf, u16* __restrict__ outb,
                            const float* __restrict__ sef, const float* __restrict__ nh,
                            const int* __restrict__ sei, int Ecnt) {
  int i = blockIdx.x * blockDim.x + threadIdx.x;
  if (i >= Ecnt * ETH) return;
  int e = i / ETH, c = i % ETH;
  float v;
  if (c < CZ) v = sef[(size_t)e * CZ + c];
  else if (c < CZ + 128) v = nh[(size_t)sei[e] * 128 + (c - CZ)];
  else v = nh[(size_t)sei[Ecnt + e] * 128 + (c - CZ - 128)];
  outf[i] = v;
  outb[i] = f2bf(v);
}

// ---------------- IPA ----------------
__global__ void k_rot_points(float* __restrict__ pts, const float* __restrict__ R,
                             const float* __restrict__ t, int rows, int P) {
  int i = blockIdx.x * blockDim.x + threadIdx.x;
  if (i >= rows * P) return;
  int n = i / P;
  float* p = pts + (size_t)i * 3;
  const float* Rm = R + (size_t)n * 9;
  float x = p[0], y = p[1], z = p[2];
  p[0] = Rm[0] * x + Rm[1] * y + Rm[2] * z + t[n * 3 + 0];
  p[1] = Rm[3] * x + Rm[4] * y + Rm[5] * z + t[n * 3 + 1];
  p[2] = Rm[6] * x + Rm[7] * y + Rm[8] * z + t[n * 3 + 2];
}

__global__ void k_ipa_logits(float* __restrict__ logits, float* __restrict__ mbuf,
                             const float* __restrict__ q, const float* __restrict__ kv,
                             const float* __restrict__ qpr, const float* __restrict__ kvpr,
                             const float* __restrict__ biasz, const float* __restrict__ gamma,
                             const float* __restrict__ mask, const int* __restrict__ ei,
                             int Ecnt) {
  int idx = blockIdx.x * blockDim.x + threadIdx.x;
  if (idx >= Ecnt * HH) return;
  int e = idx >> 3, h = idx & 7;
  int src = ei[e], dst = ei[Ecnt + e];
  const float* qv = q + (size_t)src * 512 + h * 64;
  const float* kvv = kv + (size_t)dst * 1024 + h * 128;  // k half
  float qk = 0.f;
  for (int c = 0; c < 64; ++c) qk += qv[c] * kvv[c];
  qk *= 0.07216878f;  // 1/sqrt(3*64)
  const float* qp = qpr + (size_t)src * 192 + h * 24;
  const float* kp = kvpr + (size_t)dst * 480 + h * 60;
  float d2 = 0.f;
  for (int p = 0; p < PQ; ++p) {
    float dx = qp[p * 3] - kp[p * 3];
    float dy = qp[p * 3 + 1] - kp[p * 3 + 1];
    float dz = qp[p * 3 + 2] - kp[p * 3 + 2];
    d2 += dx * dx + dy * dy + dz * dz;
  }
  float coef = log1pf(expf(gamma[h])) * 0.09622504f * -0.5f;  // sqrt(1/108)
  float lg = qk + biasz[(size_t)e * 16 + h] * 0.57735027f + coef * d2 +
             (mask[dst] - 1.f) * 100000.f;
  logits[idx] = lg;
  atomicMaxF(mbuf + src * HH + h, lg);
}

__global__ void k_ipa_w(float* __restrict__ wts, float* __restrict__ den,
                        const float* __restrict__ logits, const float* __restrict__ mbuf,
                        const int* __restrict__ ei, int Ecnt) {
  int idx = blockIdx.x * blockDim.x + threadIdx.x;
  if (idx >= Ecnt * HH) return;
  int e = idx >> 3, h = idx & 7;
  int src = ei[e];
  float w = expf(logits[idx] - mbuf[src * HH + h]);
  wts[idx] = w;
  atomicAdd(den + src * HH + h, w);
}

#define ACC_TOT (512 + 288 + 1024)
__global__ __launch_bounds__(256)
void k_ipa_accum(float* __restrict__ o, float* __restrict__ op, float* __restrict__ opair,
                 const float* __restrict__ wts, const float* __restrict__ kv,
                 const float* __restrict__ kvpr, const float* __restrict__ z,
                 const int* __restrict__ ei, int Ecnt) {
  const int e = blockIdx.x;
  const int src = ei[e], dst = ei[Ecnt + e];
  __shared__ float wsh[8];
  if (threadIdx.x < 8) wsh[threadIdx.x] = wts[(size_t)e * 8 + threadIdx.x];
  __syncthreads();
  for (int i = threadIdx.x; i < ACC_TOT; i += 256) {
    if (i < 512) {
      int h = i >> 6, c = i & 63;
      atomicAdd(o + (size_t)src * 512 + i, wsh[h] * kv[(size_t)dst * 1024 + h * 128 + 64 + c]);
    } else if (i < 800) {
      int jj = i - 512;
      int h = jj / 36, r = jj % 36;
      atomicAdd(op + (size_t)src * 288 + jj, wsh[h] * kvpr[(size_t)dst * 480 + h * 60 + 24 + r]);
    } else {
      int jj = i - 800;
      int h = jj >> 7;
      atomicAdd(opair + (size_t)src * 1024 + jj, wsh[h] * z[(size_t)e * CZ + (jj & 127)]);
    }
  }
}

__global__ __launch_bounds__(128)
void k_ipa_feats(u16* __restrict__ feats, const float* __restrict__ o,
                 const float* __restrict__ op, const float* __restrict__ opair,
                 const float* __restrict__ den, const float* __restrict__ R,
                 const float* __restrict__ t) {
  const int n = blockIdx.x;
  __shared__ float Rm[9], tv[3], sc[8];
  if (threadIdx.x < 9) Rm[threadIdx.x] = R[(size_t)n * 9 + threadIdx.x];
  if (threadIdx.x >= 16 && threadIdx.x < 19) tv[threadIdx.x - 16] = t[(size_t)n * 3 + threadIdx.x - 16];
  if (threadIdx.x >= 32 && threadIdx.x < 40)
    sc[threadIdx.x - 32] = 1.f / (den[(size_t)n * 8 + threadIdx.x - 32] + 1e-9f);
  __syncthreads();
  u16* f = feats + (size_t)n * FEATS;
  for (int i = threadIdx.x; i < 512; i += 128)
    f[i] = f2bf(o[(size_t)n * 512 + i] * sc[i >> 6]);
  for (int i = threadIdx.x; i < 96; i += 128) {
    int h = i / 12;
    float x = op[(size_t)n * 288 + i * 3 + 0] * sc[h] - tv[0];
    float y = op[(size_t)n * 288 + i * 3 + 1] * sc[h] - tv[1];
    float z = op[(size_t)n * 288 + i * 3 + 2] * sc[h] - tv[2];
    float rx = Rm[0] * x + Rm[3] * y + Rm[6] * z;
    float ry = Rm[1] * x + Rm[4] * y + Rm[7] * z;
    float rz = Rm[2] * x + Rm[5] * y + Rm[8] * z;
    f[512 + i * 3 + 0] = f2bf(rx);
    f[512 + i * 3 + 1] = f2bf(ry);
    f[512 + i * 3 + 2] = f2bf(rz);
    f[800 + i] = f2bf(sqrtf(rx * rx + ry * ry + rz * rz + 1e-8f));
  }
  for (int i = threadIdx.x; i < 1024; i += 128)
    f[896 + i] = f2bf(opair[(size_t)n * 1024 + i] * sc[i >> 7]);
}

// ---------------- frames ----------------
__global__ void k_init_frames(float* __restrict__ R, float* __restrict__ t,
                              const float* __restrict__ quats, const float* __restrict__ trans,
                              int rows) {
  int n = blockIdx.x * blockDim.x + threadIdx.x;
  if (n >= rows) return;
  const float* q = quats + (size_t)n * 4;
  float inv = rsqrtf(q[0] * q[0] + q[1] * q[1] + q[2] * q[2] + q[3] * q[3]);
  float w = q[0] * inv, x = q[1] * inv, y = q[2] * inv, z = q[3] * inv;
  float* Rn = R + (size_t)n * 9;
  Rn[0] = 1 - 2 * (y * y + z * z); Rn[1] = 2 * (x * y - w * z); Rn[2] = 2 * (x * z + w * y);
  Rn[3] = 2 * (x * y + w * z); Rn[4] = 1 - 2 * (x * x + z * z); Rn[5] = 2 * (y * z - w * x);
  Rn[6] = 2 * (x * z - w * y); Rn[7] = 2 * (y * z + w * x); Rn[8] = 1 - 2 * (x * x + y * y);
  for (int i = 0; i < 3; ++i) t[n * 3 + i] = trans[n * 3 + i];
}

__global__ void k_bb(float* __restrict__ ru, const float* __restrict__ s,
                     const float* __restrict__ w, const float* __restrict__ b,
                     const float* __restrict__ nm, int rows) {
  int idx = blockIdx.x * blockDim.x + threadIdx.x;
  if (idx >= rows * 6) return;
  int n = idx / 6, j = idx % 6;
  float m = nm[n];
  const float* sr = s + (size_t)n * CS;
  float sum = 0.f;
  for (int c = 0; c < CS; ++c) sum += sr[c] * w[c * 6 + j];
  ru[idx] = (sum * m + b[j]) * m;
}

__global__ void k_compose(float* __restrict__ R, float* __restrict__ t,
                          const float* __restrict__ ru, int rows) {
  int n = blockIdx.x * blockDim.x + threadIdx.x;
  if (n >= rows) return;
  const float* u = ru + (size_t)n * 6;
  float inv = rsqrtf(1.f + u[0] * u[0] + u[1] * u[1] + u[2] * u[2]);
  float w = inv, x = u[0] * inv, y = u[1] * inv, z = u[2] * inv;
  float Q[9] = {1 - 2 * (y * y + z * z), 2 * (x * y - w * z), 2 * (x * z + w * y),
                2 * (x * y + w * z), 1 - 2 * (x * x + z * z), 2 * (y * z - w * x),
                2 * (x * z - w * y), 2 * (y * z + w * x), 1 - 2 * (x * x + y * y)};
  float* Rn = R + (size_t)n * 9;
  float Ro[9];
  for (int i = 0; i < 3; ++i)
    for (int j = 0; j < 3; ++j) {
      float a = 0.f;
      for (int k = 0; k < 3; ++k) a += Rn[i * 3 + k] * Q[k * 3 + j];
      Ro[i * 3 + j] = a;
    }
  float tu[3];
  for (int i = 0; i < 3; ++i)
    tu[i] = t[n * 3 + i] + Rn[i * 3 + 0] * u[3] + Rn[i * 3 + 1] * u[4] + Rn[i * 3 + 2] * u[5];
  for (int i = 0; i < 9; ++i) Rn[i] = Ro[i];
  for (int i = 0; i < 3; ++i) t[n * 3 + i] = tu[i];
}

// ---------------- host side ----------------
struct Lin { const float* b; const float* w; };
struct LNp { const float* b; const float* g; };
struct IPAp { Lin bias; const float* gamma; Lin kv; Lin kvp; Lin out; Lin q; Lin qp; };
struct LayerP {
  Lin bb, ee_gate, ee_last; LNp ee_ln; Lin ee_mlp1, ee_mlp2;
  Lin et_fin, et_init; LNp et_ln; Lin et_t1, et_t2;
  IPAp ipa_seq, ipa_sp;
  Lin l2n1, l2n2, l2n3; LNp ln_s0, ln_s1, ln_s2;
  Lin n2l1, n2l2, n2l3; Lin nt1, nt2, nt3; LNp nt_ln;
  Lin se1, se2, se3; LNp se_ln;
};

static Lin pL(void* const* d, int& i) { Lin l; l.b = (const float*)d[i++]; l.w = (const float*)d[i++]; return l; }
static LNp pN(void* const* d, int& i) { LNp l; l.b = (const float*)d[i++]; l.g = (const float*)d[i++]; return l; }
static IPAp pI(void* const* d, int& i) {
  IPAp p; p.bias = pL(d, i); p.gamma = (const float*)d[i++];
  p.kv = pL(d, i); p.kvp = pL(d, i); p.out = pL(d, i); p.q = pL(d, i); p.qp = pL(d, i);
  return p;
}
static LayerP parse_layer(void* const* d, int& i) {
  LayerP L;
  L.bb = pL(d, i); L.ee_gate = pL(d, i); L.ee_last = pL(d, i); L.ee_ln = pN(d, i);
  L.ee_mlp1 = pL(d, i); L.ee_mlp2 = pL(d, i);
  L.et_fin = pL(d, i); L.et_init = pL(d, i); L.et_ln = pN(d, i);
  L.et_t1 = pL(d, i); L.et_t2 = pL(d, i);
  L.ipa_seq = pI(d, i); L.ipa_sp = pI(d, i);
  L.l2n1 = pL(d, i); L.l2n2 = pL(d, i); L.l2n3 = pL(d, i);
  L.ln_s0 = pN(d, i); L.ln_s1 = pN(d, i); L.ln_s2 = pN(d, i);
  L.n2l1 = pL(d, i); L.n2l2 = pL(d, i); L.n2l3 = pL(d, i);
  L.nt1 = pL(d, i); L.nt2 = pL(d, i); L.nt3 = pL(d, i); L.nt_ln = pN(d, i);
  L.se1 = pL(d, i); L.se2 = pL(d, i); L.se3 = pL(d, i); L.se_ln = pN(d, i);
  return L;
}

struct WS {
  float *s, *lat, *sef, *R, *t;
  u16 *s_b, *sef_b;
  u16 *arena, *wbuf;
  u16 *h1b, *h2b, *ef_b;
  float *f1, *f2, *f3, *f4;
  float *q, *kv, *qpr, *kvpr, *biasz, *logits, *wts, *m, *den, *o, *op, *opair, *upd;
  float *nh, *x0f, *eth2, *etfin;
  u16 *eth1b, *ethxb, *g1b, *g2b;
  float *g3, *m3, *ru;
};

static void setup_ws(WS& w, void* base) {
  size_t off = 0;
  auto al = [&](size_t bytes) -> void* {
    off = (off + 255) & ~(size_t)255;
    void* p = (void*)((char*)base + off);
    off += bytes;
    return p;
  };
  w.s = (float*)al((size_t)NN * CS * 4);   w.s_b = (u16*)al((size_t)NN * CS * 2);
  w.lat = (float*)al((size_t)NN * CL * 4);
  w.sef = (float*)al((size_t)ESS * CZ * 4); w.sef_b = (u16*)al((size_t)ESS * CZ * 2);
  w.R = (float*)al((size_t)NN * 9 * 4);     w.t = (float*)al((size_t)NN * 3 * 4);
  w.arena = (u16*)al((size_t)EE * EINP * 2);
  w.wbuf = (u16*)al((size_t)FEATS * 256 * 2 + 4096);
  w.h1b = (u16*)al((size_t)EE * CZ * 2);   w.h2b = (u16*)al((size_t)EE * CZ * 2);
  w.ef_b = (u16*)al((size_t)EE * CZ * 2);
  w.f1 = (float*)al((size_t)EE * CZ * 4);  w.f2 = (float*)al((size_t)EE * CZ * 4);
  w.f3 = (float*)al((size_t)EE * CZ * 4);  w.f4 = (float*)al((size_t)EE * CZ * 4);
  w.q = (float*)al((size_t)NN * 512 * 4);  w.kv = (float*)al((size_t)NN * 1024 * 4);
  w.qpr = (float*)al((size_t)NN * 192 * 4); w.kvpr = (float*)al((size_t)NN * 480 * 4);
  w.biasz = (float*)al((size_t)EE * 16 * 4);
  w.logits = (float*)al((size_t)EE * 8 * 4); w.wts = (float*)al((size_t)EE * 8 * 4);
  w.m = (float*)al((size_t)NN * 8 * 4);     w.den = (float*)al((size_t)NN * 8 * 4);
  w.o = (float*)al((size_t)NN * 512 * 4);   w.op = (float*)al((size_t)NN * 288 * 4);
  w.opair = (float*)al((size_t)NN * 1024 * 4);
  w.upd = (float*)al((size_t)NN * 256 * 4);
  w.nh = (float*)al((size_t)NN * 128 * 4);
  w.x0f = (float*)al((size_t)ESS * ETH * 4);
  w.eth2 = (float*)al((size_t)ESS * ETH * 4);
  w.etfin = (float*)al((size_t)ESS * CZ * 4);
  w.eth1b = (u16*)al((size_t)ESS * ETH * 2);
  w.ethxb = (u16*)al((size_t)ESS * ETH * 2);
  w.g1b = (u16*)al((size_t)NN * 256 * 2);   w.g2b = (u16*)al((size_t)NN * 256 * 2);
  w.g3 = (float*)al((size_t)NN * 256 * 4);
  w.m3 = (float*)al((size_t)NN * 128 * 4);
  w.ru = (float*)al((size_t)NN * 6 * 4);
}

static inline int cdiv(int a, int b) { return (a + b - 1) / b; }

static void gemm(hipStream_t st, WS& ws, const u16* A, int M, int K, int Kp,
                 const float* w, const float* b, int Nc, int Ncp,
                 float* outf, u16* outb, int act) {
  int tot = Kp * Ncp;
  k_w2bf<<<cdiv(tot, 256), 256, 0, st>>>(w, ws.wbuf, K, Kp, Nc, Ncp);
  dim3 grid(M / 32, cdiv(Ncp, 128));
  k_gemm<<<grid, 128, 0, st>>>(A, ws.wbuf, b, Nc, outf, outb, Kp, Ncp, act);
}

static void run_ipa(hipStream_t st, WS& ws, const IPAp& p, const int* ei, int Ecnt,
                    const float* z, const u16* z_b, const float* mask) {
  gemm(st, ws, ws.s_b, NN, CS, CS, p.q.w, p.q.b, 512, 512, ws.q, nullptr, 0);
  gemm(st, ws, ws.s_b, NN, CS, CS, p.kv.w, p.kv.b, 1024, 1024, ws.kv, nullptr, 0);
  gemm(st, ws, ws.s_b, NN, CS, CS, p.qp.w, p.qp.b, 192, 192, ws.qpr, nullptr, 0);
  k_rot_points<<<cdiv(NN * 64, 256), 256, 0, st>>>(ws.qpr, ws.R, ws.t, NN, 64);
  gemm(st, ws, ws.s_b, NN, CS, CS, p.kvp.w, p.kvp.b, 480, 480, ws.kvpr, nullptr, 0);
  k_rot_points<<<cdiv(NN * 160, 256), 256, 0, st>>>(ws.kvpr, ws.R, ws.t, NN, 160);
  gemm(st, ws, z_b, Ecnt, CZ, CZ, p.bias.w, p.bias.b, 8, 16, ws.biasz, nullptr, 0);
  k_fill<<<cdiv(NN * 8, 256), 256, 0, st>>>(ws.m, -1e30f, NN * 8);
  k_fill<<<cdiv(NN * 8, 256), 256, 0, st>>>(ws.den, 0.f, NN * 8);
  k_fill<<<cdiv(NN * 512, 256), 256, 0, st>>>(ws.o, 0.f, NN * 512);
  k_fill<<<cdiv(NN * 288, 256), 256, 0, st>>>(ws.op, 0.f, NN * 288);
  k_fill<<<cdiv(NN * 1024, 256), 256, 0, st>>>(ws.opair, 0.f, NN * 1024);
  k_ipa_logits<<<cdiv(Ecnt * 8, 256), 256, 0, st>>>(ws.logits, ws.m, ws.q, ws.kv, ws.qpr,
                                                    ws.kvpr, ws.biasz, p.gamma, mask, ei, Ecnt);
  k_ipa_w<<<cdiv(Ecnt * 8, 256), 256, 0, st>>>(ws.wts, ws.den, ws.logits, ws.m, ei, Ecnt);
  k_ipa_accum<<<Ecnt, 256, 0, st>>>(ws.o, ws.op, ws.opair, ws.wts, ws.kv, ws.kvpr, z, ei, Ecnt);
  k_ipa_feats<<<NN, 128, 0, st>>>(ws.arena, ws.o, ws.op, ws.opair, ws.den, ws.R, ws.t);
  gemm(st, ws, ws.arena, NN, FEATS, FEATS, p.out.w, p.out.b, CS, CS, ws.upd, nullptr, 0);
}

static void run_layer(hipStream_t st, WS& ws, const LayerP& L,
                      const float* edge_raw, const int* ei,
                      const float* nsei_raw, const int* sei,
                      const float* res_mask, const float* seq_mask,
                      const float* res_nm, const float* seq_nm) {
  // 1. ee edge MLP -> ef
  k_gather_edge<<<EE, 128, 0, st>>>(ws.arena, edge_raw, ws.s, ws.lat, ei, EE);
  gemm(st, ws, ws.arena, EE, EIN, EINP, L.ee_mlp1.w, L.ee_mlp1.b, CZ, CZ, nullptr, ws.h1b, 1);
  gemm(st, ws, ws.h1b, EE, CZ, CZ, L.ee_mlp2.w, L.ee_mlp2.b, CZ, CZ, nullptr, ws.h2b, 1);
  gemm(st, ws, ws.h2b, EE, CZ, CZ, L.ee_gate.w, L.ee_gate.b, CZ, CZ, ws.f1, nullptr, 0);
  gemm(st, ws, ws.h2b, EE, CZ, CZ, L.ee_last.w, L.ee_last.b, CZ, CZ, ws.f2, nullptr, 0);
  k_sigmul<<<cdiv(EE * CZ, 256), 256, 0, st>>>(ws.f3, ws.f1, ws.f2, EE * CZ);
  k_layernorm<<<EE, 128, 0, st>>>(ws.f3, nullptr, nullptr, L.ee_ln.g, L.ee_ln.b, ws.f4, ws.ef_b, CZ);

  // 2. se edge MLP -> sef
  k_gather_edge<<<ESS, 128, 0, st>>>(ws.arena, nsei_raw, ws.s, ws.lat, sei, ESS);
  gemm(st, ws, ws.arena, ESS, EIN, EINP, L.se1.w, L.se1.b, CZ, CZ, nullptr, ws.h1b, 1);
  gemm(st, ws, ws.h1b, ESS, CZ, CZ, L.se2.w, L.se2.b, CZ, CZ, nullptr, ws.h2b, 1);
  gemm(st, ws, ws.h2b, ESS, CZ, CZ, L.se3.w, L.se3.b, CZ, CZ, ws.f1, nullptr, 0);
  k_layernorm<<<ESS, 128, 0, st>>>(ws.sef, ws.f1, nullptr, L.se_ln.g, L.se_ln.b, ws.sef, ws.sef_b, CZ);

  // 3. l2n: s = ln_s0(s + ff3([lat|s]) * seq_mask)
  k_concat_ls<<<cdiv(NN * 384, 256), 256, 0, st>>>(ws.arena, ws.lat, ws.s, nullptr, NN);
  gemm(st, ws, ws.arena, NN, 384, 384, L.l2n1.w, L.l2n1.b, CS, CS, nullptr, ws.g1b, 1);
  gemm(st, ws, ws.g1b, NN, CS, CS, L.l2n2.w, L.l2n2.b, CS, CS, nullptr, ws.g2b, 1);
  gemm(st, ws, ws.g2b, NN, CS, CS, L.l2n3.w, L.l2n3.b, CS, CS, ws.g3, nullptr, 0);
  k_layernorm<<<NN, 128, 0, st>>>(ws.s, ws.g3, seq_mask, L.ln_s0.g, L.ln_s0.b, ws.s, ws.s_b, CS);

  // 4. spatial IPA
  run_ipa(st, ws, L.ipa_sp, ei, EE, ws.f4, ws.ef_b, res_mask);
  k_layernorm<<<NN, 128, 0, st>>>(ws.s, ws.upd, res_mask, L.ln_s1.g, L.ln_s1.b, ws.s, ws.s_b, CS);

  // 5. sequence IPA
  run_ipa(st, ws, L.ipa_seq, sei, ESS, ws.sef, ws.sef_b, res_mask);
  k_layernorm<<<NN, 128, 0, st>>>(ws.s, ws.upd, res_mask, L.ln_s2.g, L.ln_s2.b, ws.s, ws.s_b, CS);

  // 6. node transition
  gemm(st, ws, ws.s_b, NN, CS, CS, L.nt1.w, L.nt1.b, CS, CS, nullptr, ws.g1b, 1);
  gemm(st, ws, ws.g1b, NN, CS, CS, L.nt2.w, L.nt2.b, CS, CS, nullptr, ws.g2b, 1);
  gemm(st, ws, ws.g2b, NN, CS, CS, L.nt3.w, L.nt3.b, CS, CS, ws.g3, nullptr, 0);
  k_layernorm<<<NN, 128, 0, st>>>(ws.s, ws.g3, nullptr, L.nt_ln.g, L.nt_ln.b, ws.s, ws.s_b, CS);
  k_rowscale_dual<<<cdiv(NN * CS, 256), 256, 0, st>>>(ws.s, ws.s_b, res_mask, NN, CS);

  // 7. backbone update
  k_bb<<<cdiv(NN * 6, 256), 256, 0, st>>>(ws.ru, ws.s, L.bb.w, L.bb.b, res_nm, NN);
  k_compose<<<cdiv(NN, 256), 256, 0, st>>>(ws.R, ws.t, ws.ru, NN);

  // 8. edge transformer -> sef
  gemm(st, ws, ws.s_b, NN, CS, CS, L.et_init.w, L.et_init.b, 128, 128, ws.nh, nullptr, 0);
  k_concat_et<<<cdiv(ESS * ETH, 256), 256, 0, st>>>(ws.x0f, ws.arena, ws.sef, ws.nh, sei, ESS);
  gemm(st, ws, ws.arena, ESS, ETH, ETH, L.et_t1.w, L.et_t1.b, ETH, ETH, nullptr, ws.eth1b, 1);
  gemm(st, ws, ws.eth1b, ESS, ETH, ETH, L.et_t2.w, L.et_t2.b, ETH, ETH, ws.eth2, nullptr, 1);
  k_addcvt<<<cdiv(ESS * ETH, 256), 256, 0, st>>>(ws.ethxb, ws.eth2, ws.x0f, ESS * ETH);
  gemm(st, ws, ws.ethxb, ESS, ETH, ETH, L.et_fin.w, L.et_fin.b, CZ, CZ, ws.etfin, nullptr, 0);
  k_layernorm<<<ESS, 128, 0, st>>>(ws.etfin, nullptr, nullptr, L.et_ln.g, L.et_ln.b, ws.sef, ws.sef_b, CZ);

  // 9. n2l: lat += ff3([lat|s]*seq_mask) * seq_nm
  k_concat_ls<<<cdiv(NN * 384, 256), 256, 0, st>>>(ws.arena, ws.lat, ws.s, seq_mask, NN);
  gemm(st, ws, ws.arena, NN, 384, 384, L.n2l1.w, L.n2l1.b, CL, CL, nullptr, ws.h1b, 1);
  gemm(st, ws, ws.h1b, NN, CL, CL, L.n2l2.w, L.n2l2.b, CL, CL, nullptr, ws.h2b, 1);
  gemm(st, ws, ws.h2b, NN, CL, CL, L.n2l3.w, L.n2l3.b, CL, CL, ws.m3, nullptr, 0);
  k_add_rowscale<<<cdiv(NN * CL, 256), 256, 0, st>>>(ws.lat, ws.m3, seq_nm, NN, CL);
}

extern "C" void kernel_launch(void* const* d_in, const int* in_sizes, int n_in,
                              void* d_out, int out_size, void* d_ws, size_t ws_size,
                              hipStream_t stream) {
  (void)in_sizes; (void)n_in; (void)out_size; (void)ws_size;
  const float* node_features = (const float*)d_in[0];
  const float* quats = (const float*)d_in[1];
  const float* trans = (const float*)d_in[2];
  const float* latent = (const float*)d_in[3];
  const float* edge_raw = (const float*)d_in[4];
  const int* ei = (const int*)d_in[5];
  const float* nsei_raw = (const float*)d_in[6];
  const float* sef0 = (const float*)d_in[7];
  const int* sei = (const int*)d_in[8];
  const float* res_mask = (const float*)d_in[9];
  const float* seq_mask = (const float*)d_in[10];
  const float* res_nm = (const float*)d_in[11];
  const float* seq_nm = (const float*)d_in[12];

  int idx = 13;
  LayerP layers[2];
  for (int l = 0; l < 2; ++l) layers[l] = parse_layer(d_in, idx);

  WS ws;
  setup_ws(ws, d_ws);

  k_copy_dual<<<cdiv(NN * CS, 256), 256, 0, stream>>>(ws.s, ws.s_b, node_features, NN * CS);
  k_copy_dual<<<cdiv(NN * CL, 256), 256, 0, stream>>>(ws.lat, (u16*)nullptr, latent, NN * CL);
  k_copy_dual<<<cdiv(ESS * CZ, 256), 256, 0, stream>>>(ws.sef, ws.sef_b, sef0, ESS * CZ);
  k_init_frames<<<cdiv(NN, 256), 256, 0, stream>>>(ws.R, ws.t, quats, trans, NN);

  for (int l = 0; l < 2; ++l)
    run_layer(stream, ws, layers[l], edge_raw, ei, nsei_raw, sei,
              res_mask, seq_mask, res_nm, seq_nm);

  k_copy_dual<<<cdiv(NN * CS, 256), 256, 0, stream>>>((float*)d_out, (u16*)nullptr, ws.s, NN * CS);
}
